// Series_85658827751552
// MI455X (gfx1250) — matrix operand order repaired and checked by reading, not compiled, not run on hardware
//
#include <hip/hip_runtime.h>
#include <hip/hip_bf16.h>

typedef __attribute__((ext_vector_type(16))) __bf16 v16bf;
typedef __attribute__((ext_vector_type(8)))  float  v8f;

// Sizes: T=256, B=1024, S=64, H=128, 4H=512
// Batch tile per block = 32 rows (2 M-tiles), 32 blocks, 256 threads (8 waves).
// Wave w owns gate tiles {w, w+8, w+16, w+24} = i,f,g,o for hidden cols
// [w*16, w*16+16), so the LSTM cell update is fully register-resident.

// ---------------------------------------------------------------------------
// Prep: pack combined weights into per-lane WMMA B-fragment order (bf16),
// and fold biases.  Layout per 32x16 (KxN) tile:
//   lane l (0..31), element e (0..15):
//     n = l & 15;  K = (l < 16) ? e : 16 + e
//   stored contiguously: ((kc*32 + nt)*32 + lane)*16 + e
// Layer0 combined W: [512 x 192] = [ w_ih0 @ pre_w (512x64) | w_hh0 (512x128) ]
// Layer1 combined W: [512 x 256] = [ w_ih1 | w_hh1 ]
// bias0 = w_ih0 @ pre_b + b_ih0 + b_hh0 ;  bias1 = b_ih1 + b_hh1
// ---------------------------------------------------------------------------
__global__ void prep_kernel(const float* __restrict__ pre_w, const float* __restrict__ pre_b,
                            const float* __restrict__ w_ih0, const float* __restrict__ w_hh0,
                            const float* __restrict__ b_ih0, const float* __restrict__ b_hh0,
                            const float* __restrict__ w_ih1, const float* __restrict__ w_hh1,
                            const float* __restrict__ b_ih1, const float* __restrict__ b_hh1,
                            __bf16* __restrict__ W0p, __bf16* __restrict__ W1p,
                            float* __restrict__ bias0, float* __restrict__ bias1)
{
    const int N0 = 6 * 32 * 512;   // layer0 packed elements (KC=6)
    const int N1 = 8 * 32 * 512;   // layer1 packed elements (KC=8)
    int p = blockIdx.x * blockDim.x + threadIdx.x;
    if (p < N0) {
        int tile = p >> 9, w = p & 511, lane = w >> 4, e = w & 15;
        int kc = tile >> 5, nt = tile & 31;
        int row = nt * 16 + (lane & 15);
        int K   = kc * 32 + ((lane < 16) ? e : 16 + e);
        float v;
        if (K < 64) {  // (w_ih0 @ pre_w)[row, K]
            float acc = 0.f;
            for (int h = 0; h < 128; ++h) acc += w_ih0[row * 128 + h] * pre_w[h * 64 + K];
            v = acc;
        } else {
            v = w_hh0[row * 128 + (K - 64)];
        }
        W0p[p] = (__bf16)v;
    } else if (p < N0 + N1) {
        int q = p - N0;
        int tile = q >> 9, w = q & 511, lane = w >> 4, e = w & 15;
        int kc = tile >> 5, nt = tile & 31;
        int row = nt * 16 + (lane & 15);
        int K   = kc * 32 + ((lane < 16) ? e : 16 + e);
        float v = (K < 128) ? w_ih1[row * 128 + K] : w_hh1[row * 128 + (K - 128)];
        W1p[q] = (__bf16)v;
    } else if (p < N0 + N1 + 1024) {
        int n = p - N0 - N1;
        if (n < 512) {
            float acc = b_ih0[n] + b_hh0[n];
            for (int h = 0; h < 128; ++h) acc += w_ih0[n * 128 + h] * pre_b[h];
            bias0[n] = acc;
        } else {
            int m = n - 512;
            bias1[m] = b_ih1[m] + b_hh1[m];
        }
    }
}

// ---------------------------------------------------------------------------
// One LSTM layer over all T steps for a 32-row batch tile.
// ---------------------------------------------------------------------------
template <int KC, int XOFF, bool LAYER0>
__global__ __launch_bounds__(256) void lstm_pass_kernel(
    const float*  __restrict__ x_in,   // [T,B,64]  (layer0)
    const __bf16* __restrict__ y_in,   // [T,B,128] bf16 (layer1 input)
    __bf16*       __restrict__ y_out,  // [T,B,128] bf16 (layer0 output)
    const __bf16* __restrict__ Wp,     // packed [KC][32 tiles][512]
    const float*  __restrict__ bias,   // [512] folded bias
    const float*  __restrict__ h0,     // [B,128] layer slice
    const float*  __restrict__ c0,     // [B,128] layer slice
    float* __restrict__ hT, float* __restrict__ cT)
{
    __shared__ __bf16 sW[KC * 32 * 512];  // 192/256 KB packed weight panels
    __shared__ __bf16 sA[32][256];        // 16 KB A panel: [x_t | h] per row

    const int tid   = threadIdx.x;
    const int lane  = tid & 31;
    const int wave  = tid >> 5;
    const int brow0 = blockIdx.x * 32;

    // Stage packed weights into LDS once (reused for all 256 steps).
    {
        const int n16 = KC * 2048;  // KC*32*512*2 bytes / 16
        const uint4* src = (const uint4*)Wp;
        uint4* dst = (uint4*)sW;
        for (int i = tid; i < n16; i += 256) dst[i] = src[i];
    }

    // This lane's hidden column and row set (matches WMMA C layout):
    //   j = wave*16 + (lane&15);  rows m = mt*16 + rbase + r
    const int jj    = wave * 16 + (lane & 15);
    const int rbase = (lane >> 4) * 8;

    // Persistent per-lane cell/hidden state: 16 elements (2 M-tiles x 8 rows).
    float creg[16], hreg[16];
#pragma unroll
    for (int q = 0; q < 16; ++q) {
        int mt = q >> 3, r = q & 7;
        int m = mt * 16 + rbase + r;
        size_t gi = (size_t)(brow0 + m) * 128 + jj;
        creg[q] = c0[gi];
        float hv = h0[gi];
        hreg[q] = hv;
        sA[m][XOFF + jj] = (__bf16)hv;
    }

    // Folded bias for this wave's 4 gate tiles {w, w+8, w+16, w+24}.
    float bv[4];
#pragma unroll
    for (int nt = 0; nt < 4; ++nt) bv[nt] = bias[(wave + nt * 8) * 16 + (lane & 15)];

    const int mA   = lane & 15;
    const int offA = (lane >> 4) * 16;  // A-frag K offset: contiguous K = 16*half + e

    union Frag { v16bf v; uint4 u[2]; };

    for (int t = 0; t < 256; ++t) {
        // Stage this step's input slice into A panel columns [0, XOFF).
        if (LAYER0) {
#pragma unroll
            for (int r = 0; r < 8; ++r) {           // 32*64 elems / 256 threads
                int i = tid + r * 256;
                int m = i >> 6, c = i & 63;
                sA[m][c] = (__bf16)x_in[((size_t)t * 1024 + brow0 + m) * 64 + c];
            }
            if (t + 1 < 256)
                __builtin_prefetch(&x_in[((size_t)(t + 1) * 1024 + brow0) * 64], 0, 1);
        } else {
            // copy 32*128 bf16 as 32-bit words
            uint* dstu = (uint*)&sA[0][0];
            const uint* srcu = (const uint*)y_in;
#pragma unroll
            for (int r = 0; r < 8; ++r) {           // 2048 words / 256 threads
                int i = tid + r * 256;
                int m = i >> 6, cu = i & 63;
                dstu[m * 128 + cu] = srcu[((size_t)t * 1024 + brow0 + m) * 64 + cu];
            }
            if (t + 1 < 256)
                __builtin_prefetch(&y_in[((size_t)(t + 1) * 1024 + brow0) * 128], 0, 1);
        }
        __syncthreads();  // x_t + h(t-1) visible to all waves

        // GEMM: gates[32,512] = [x_t | h] @ Wc.T + bias
        v8f acc[4][2];
#pragma unroll
        for (int nt = 0; nt < 4; ++nt) {
            float b = bv[nt];
            v8f ini = {b, b, b, b, b, b, b, b};
            acc[nt][0] = ini;
            acc[nt][1] = ini;
        }

#pragma unroll
        for (int kc = 0; kc < KC; ++kc) {
            Frag a0, a1;
            a0.u[0] = *(const uint4*)&sA[mA][kc * 32 + offA];
            a0.u[1] = *(const uint4*)&sA[mA][kc * 32 + offA + 8];
            a1.u[0] = *(const uint4*)&sA[16 + mA][kc * 32 + offA];
            a1.u[1] = *(const uint4*)&sA[16 + mA][kc * 32 + offA + 8];
#pragma unroll
            for (int nt = 0; nt < 4; ++nt) {
                int ntg = wave + nt * 8;
                Frag bF;
                const uint4* p = (const uint4*)&sW[((size_t)(kc * 32 + ntg) * 32 + lane) * 16];
                bF.u[0] = p[0];
                bF.u[1] = p[1];
                acc[nt][0] = __builtin_amdgcn_wmma_f32_16x16x32_bf16(
                    false, a0.v, false, bF.v, (short)0, acc[nt][0], false, false);
                acc[nt][1] = __builtin_amdgcn_wmma_f32_16x16x32_bf16(
                    false, a1.v, false, bF.v, (short)0, acc[nt][1], false, false);
            }
        }
        __syncthreads();  // all A-panel reads done before h is overwritten

        // Register-resident LSTM cell update: acc[0..3] = i,f,g,o for (m, jj).
#pragma unroll
        for (int mt = 0; mt < 2; ++mt) {
#pragma unroll
            for (int r = 0; r < 8; ++r) {
                int q = mt * 8 + r;
                float ig = acc[0][mt][r];
                float fg = acc[1][mt][r];
                float gg = acc[2][mt][r];
                float og = acc[3][mt][r];
                float i_s = 1.f / (1.f + __expf(-ig));
                float f_s = 1.f / (1.f + __expf(-fg));
                float g_t = tanhf(gg);
                float o_s = 1.f / (1.f + __expf(-og));
                float c = f_s * creg[q] + i_s * g_t;
                creg[q] = c;
                float h = o_s * tanhf(c);
                hreg[q] = h;
                int m = mt * 16 + rbase + r;
                sA[m][XOFF + jj] = (__bf16)h;
                if (LAYER0)
                    y_out[((size_t)t * 1024 + brow0 + m) * 128 + jj] = (__bf16)h;
            }
        }
        // barrier at top of next iteration orders h-writes vs. A-frag reads
    }

#pragma unroll
    for (int q = 0; q < 16; ++q) {
        int mt = q >> 3, r = q & 7;
        int m = mt * 16 + rbase + r;
        size_t gi = (size_t)(brow0 + m) * 128 + jj;
        hT[gi] = hreg[q];
        cT[gi] = creg[q];
    }
}

// ---------------------------------------------------------------------------
// post: pred[b] = relu(h1_last[b,:]) @ post_w + post_b
// ---------------------------------------------------------------------------
__global__ void post_kernel(const float* __restrict__ h1,
                            const float* __restrict__ post_w,
                            const float* __restrict__ post_b,
                            float* __restrict__ pred)
{
    int b = blockIdx.x * blockDim.x + threadIdx.x;
    if (b < 1024) {
        float acc = post_b[0];
        for (int j = 0; j < 128; ++j) {
            float h = h1[b * 128 + j];
            h = h > 0.f ? h : 0.f;
            acc += h * post_w[j];
        }
        pred[b] = acc;
    }
}

// ---------------------------------------------------------------------------
extern "C" void kernel_launch(void* const* d_in, const int* in_sizes, int n_in,
                              void* d_out, int out_size, void* d_ws, size_t ws_size,
                              hipStream_t stream)
{
    const float* x_in   = (const float*)d_in[0];
    const float* h0     = (const float*)d_in[1];
    const float* c0     = (const float*)d_in[2];
    const float* pre_w  = (const float*)d_in[3];
    const float* pre_b  = (const float*)d_in[4];
    const float* w_ih0  = (const float*)d_in[5];
    const float* w_hh0  = (const float*)d_in[6];
    const float* b_ih0  = (const float*)d_in[7];
    const float* b_hh0  = (const float*)d_in[8];
    const float* w_ih1  = (const float*)d_in[9];
    const float* w_hh1  = (const float*)d_in[10];
    const float* b_ih1  = (const float*)d_in[11];
    const float* b_hh1  = (const float*)d_in[12];
    const float* post_w = (const float*)d_in[13];
    const float* post_b = (const float*)d_in[14];

    // Output: pred [1024], h_t [2,1024,128], c_t [2,1024,128] (flat concat)
    float* out  = (float*)d_out;
    float* pred = out;
    float* hT0  = out + 1024;
    float* hT1  = hT0 + 1024 * 128;
    float* cT0  = hT1 + 1024 * 128;
    float* cT1  = cT0 + 1024 * 128;

    // Workspace layout
    char* ws = (char*)d_ws;
    __bf16* y0   = (__bf16*)ws;  ws += (size_t)256 * 1024 * 128 * 2;  // 64 MB
    __bf16* W0p  = (__bf16*)ws;  ws += (size_t)6 * 32 * 512 * 2;
    __bf16* W1p  = (__bf16*)ws;  ws += (size_t)8 * 32 * 512 * 2;
    float*  bias0 = (float*)ws;  ws += 512 * 4;
    float*  bias1 = (float*)ws;

    {
        int total = 6 * 32 * 512 + 8 * 32 * 512 + 1024;
        prep_kernel<<<(total + 255) / 256, 256, 0, stream>>>(
            pre_w, pre_b, w_ih0, w_hh0, b_ih0, b_hh0,
            w_ih1, w_hh1, b_ih1, b_hh1, W0p, W1p, bias0, bias1);
    }

    // Layer 0: input = x_in (pre-linear folded), K = 64+128 = 192 (KC=6)
    lstm_pass_kernel<6, 64, true><<<32, 256, 0, stream>>>(
        x_in, nullptr, y0, W0p, bias0,
        h0, c0, hT0, cT0);

    // Layer 1: input = y0 (bf16), K = 128+128 = 256 (KC=8)
    lstm_pass_kernel<8, 128, false><<<32, 256, 0, stream>>>(
        nullptr, y0, nullptr, W1p, bias1,
        h0 + 1024 * 128, c0 + 1024 * 128, hT1, cT1);

    post_kernel<<<4, 256, 0, stream>>>(hT1, post_w, post_b, pred);
}
